// PyGTGNN_56221121905002
// MI455X (gfx1250) — compile-verified, matrix operand order checked by reading, not run on hardware
//
#include <hip/hip_runtime.h>

#define T_STEPS 12
#define F_IN 16
#define H 64

typedef __attribute__((ext_vector_type(16))) _Float16 v16h;
typedef __attribute__((ext_vector_type(8)))  _Float16 v8h;
typedef __attribute__((ext_vector_type(8)))  float    v8f;

// ---------------------------------------------------------------------------
// A/B fragment loader for v_wmma_f32_16x16x32_f16.
// Per ISA (CDNA5 ISA §7.12.2, 16-bit A 16x32):
//   lanes 0-15 : halves[0..7]=K(kchunk+0..7),  halves[8..15]=K(kchunk+16..23)
//   lanes 16-31: halves[0..7]=K(kchunk+8..15), halves[8..15]=K(kchunk+24..31)
// B (32x16, K x N) mirrors this with lane = column index n (row striped
// across lanes within one VGPR), so a B fragment for computing a @ W^T is a
// contiguous slice of row n of the row-major weight matrix W[192,64].
// kb below = (lane<16 ? 0 : 8) + 32*chunk.
// ---------------------------------------------------------------------------
__device__ __forceinline__ v16h load_fragA(const _Float16* __restrict__ row, int kb) {
  union { v16h v; v8h h[2]; } u;
  u.h[0] = *(const v8h*)(row + kb);
  u.h[1] = *(const v8h*)(row + kb + 16);
  return u.v;
}

// Branch-free fast transcendentals: single v_exp_f32 + v_rcp_f32, no IEEE
// division sequence, no EXEC-mask divergence inside the WMMA loop.
__device__ __forceinline__ float sigmoidf_(float x) {
  return __builtin_amdgcn_rcpf(1.0f + __expf(-x));
}
__device__ __forceinline__ float tanhf_(float x) {
  // tanh(x) = 1 - 2/(exp(2x)+1); exp->inf => 1, exp->0 => -1 (correct limits)
  return 1.0f - 2.0f * __builtin_amdgcn_rcpf(1.0f + __expf(2.0f * x));
}

// ------------------------- one-time setup kernels --------------------------

__global__ __launch_bounds__(256) void init_deg_kernel(float* __restrict__ deg, int N) {
  int i = blockIdx.x * 256 + threadIdx.x;
  if (i < N) deg[i] = 1.0f;  // self-loop contribution
}

__global__ __launch_bounds__(256) void deg_accum_kernel(const int* __restrict__ dst,
                                                        float* __restrict__ deg, int E) {
  int e = blockIdx.x * 256 + threadIdx.x;
  if (e < E) atomicAdd(&deg[dst[e]], 1.0f);
}

__global__ __launch_bounds__(256) void dinv_kernel(float* __restrict__ deg, int N) {
  int i = blockIdx.x * 256 + threadIdx.x;
  if (i < N) deg[i] = rsqrtf(deg[i]);  // deg >= 1 always (self-loop)
}

__global__ __launch_bounds__(256) void edge_w_kernel(const int* __restrict__ src,
                                                     const int* __restrict__ dst,
                                                     const float* __restrict__ dinv,
                                                     float* __restrict__ w, int E) {
  int e = blockIdx.x * 256 + threadIdx.x;
  if (e < E) w[e] = dinv[src[e]] * dinv[dst[e]];
}

__global__ __launch_bounds__(256) void cvt_weights_kernel(const float* __restrict__ Wih,
                                                          const float* __restrict__ Whh,
                                                          _Float16* __restrict__ Wih16,
                                                          _Float16* __restrict__ Whh16,
                                                          int n) {
  int i = blockIdx.x * 256 + threadIdx.x;
  if (i < n) {
    Wih16[i] = (_Float16)Wih[i];
    Whh16[i] = (_Float16)Whh[i];
  }
}

__global__ __launch_bounds__(256) void init_h_kernel(float* __restrict__ h,
                                                     _Float16* __restrict__ h16, int n) {
  int i = blockIdx.x * 256 + threadIdx.x;
  if (i < n) { h[i] = 0.0f; h16[i] = (_Float16)0.0f; }
}

// ------------------------- per-timestep kernels ----------------------------

// g = X[:,t,:] @ Wg ; agg initialized with self-loop term dinv^2 * g
__global__ __launch_bounds__(256) void transform_kernel(const float* __restrict__ X,
                                                        const float* __restrict__ Wg,
                                                        const float* __restrict__ dinv,
                                                        float* __restrict__ g,
                                                        float* __restrict__ agg,
                                                        int t, int N) {
  __shared__ float sW[F_IN * H];   // 4 KB
  __shared__ float sX[4 * F_IN];
  int tid = threadIdx.x;
  #pragma unroll
  for (int i = tid; i < F_IN * H; i += 256) sW[i] = Wg[i];
  if (tid < 64) {
    int ni = blockIdx.x * 4 + (tid >> 4);
    int k  = tid & 15;
    sX[tid] = (ni < N) ? X[((size_t)ni * T_STEPS + t) * F_IN + k] : 0.0f;
  }
  __syncthreads();
  int node = blockIdx.x * 4 + (tid >> 6);
  int hc   = tid & 63;
  if (node < N) {
    const float* xr = &sX[(tid >> 6) * F_IN];
    float acc = 0.0f;
    #pragma unroll
    for (int k = 0; k < F_IN; ++k) acc += xr[k] * sW[k * H + hc];
    float di = dinv[node];
    size_t idx = (size_t)node * H + hc;
    g[idx]   = acc;
    agg[idx] = acc * di * di;
  }
}

// agg[dst] += w_e * g[src]  (16 threads per edge, float4 gathers)
__global__ __launch_bounds__(256) void scatter_kernel(const int* __restrict__ src,
                                                      const int* __restrict__ dst,
                                                      const float* __restrict__ w,
                                                      const float* __restrict__ g,
                                                      float* __restrict__ agg, int E) {
  int tid = blockIdx.x * 256 + threadIdx.x;
  int e = tid >> 4;
  if (e >= E) return;
  int q = tid & 15;
  int s = src[e], d = dst[e];
  float we = w[e];
  const float4 gv = *(const float4*)(g + (size_t)s * H + q * 4);
  float* ap = agg + (size_t)d * H + q * 4;
  atomicAdd(ap + 0, we * gv.x);
  atomicAdd(ap + 1, we * gv.y);
  atomicAdd(ap + 2, we * gv.z);
  atomicAdd(ap + 3, we * gv.w);
}

// a16 = f16(relu(agg + bg))
__global__ __launch_bounds__(256) void act_kernel(const float* __restrict__ agg,
                                                  const float* __restrict__ bg,
                                                  _Float16* __restrict__ a16, size_t n) {
  size_t i = (size_t)blockIdx.x * 256 + threadIdx.x;
  if (i < n) a16[i] = (_Float16)fmaxf(agg[i] + bg[i & 63], 0.0f);
}

// GRU cell: gi = a @ W_ih^T + b_ih ; gh = h @ W_hh^T + b_hh ; gates -> h_new.
// One wave32 per 16-node tile; 48 v_wmma_f32_16x16x32_f16 per tile.
__global__ __launch_bounds__(32) void gru_wmma_kernel(const _Float16* __restrict__ a16,
                                                      _Float16* h16,
                                                      const float* __restrict__ hcur,
                                                      float* __restrict__ hnxt,
                                                      const _Float16* __restrict__ Wih16,
                                                      const _Float16* __restrict__ Whh16,
                                                      const float* __restrict__ b_ih,
                                                      const float* __restrict__ b_hh) {
  const int lane  = threadIdx.x;
  const int node0 = blockIdx.x * 16;
  const int m     = lane & 15;
  const int kb    = (lane < 16) ? 0 : 8;
  const int mrow0 = (lane >> 4) << 3;  // C/D layout: lanes 16-31 hold rows M=8..15

  const _Float16* arow = a16 + (size_t)(node0 + m) * H;
  const _Float16* hrow = h16 + (size_t)(node0 + m) * H;
  v16h aA0 = load_fragA(arow, kb);
  v16h aA1 = load_fragA(arow, 32 + kb);
  v16h hA0 = load_fragA(hrow, kb);
  v16h hA1 = load_fragA(hrow, 32 + kb);

  #pragma unroll
  for (int j = 0; j < 4; ++j) {
    const _Float16* wr = Wih16 + (size_t)(      j * 16 + m) * H;
    const _Float16* wz = Wih16 + (size_t)( 64 + j * 16 + m) * H;
    const _Float16* wn = Wih16 + (size_t)(128 + j * 16 + m) * H;
    const _Float16* ur = Whh16 + (size_t)(      j * 16 + m) * H;
    const _Float16* uz = Whh16 + (size_t)( 64 + j * 16 + m) * H;
    const _Float16* un = Whh16 + (size_t)(128 + j * 16 + m) * H;

    v8f ir = {}, iz = {}, in_ = {}, hr = {}, hz = {}, hn = {};

    // K chunk 0 (K = 0..31)
    ir  = __builtin_amdgcn_wmma_f32_16x16x32_f16(false, aA0, false, load_fragA(wr, kb), (short)0, ir,  false, false);
    iz  = __builtin_amdgcn_wmma_f32_16x16x32_f16(false, aA0, false, load_fragA(wz, kb), (short)0, iz,  false, false);
    in_ = __builtin_amdgcn_wmma_f32_16x16x32_f16(false, aA0, false, load_fragA(wn, kb), (short)0, in_, false, false);
    hr  = __builtin_amdgcn_wmma_f32_16x16x32_f16(false, hA0, false, load_fragA(ur, kb), (short)0, hr,  false, false);
    hz  = __builtin_amdgcn_wmma_f32_16x16x32_f16(false, hA0, false, load_fragA(uz, kb), (short)0, hz,  false, false);
    hn  = __builtin_amdgcn_wmma_f32_16x16x32_f16(false, hA0, false, load_fragA(un, kb), (short)0, hn,  false, false);
    // K chunk 1 (K = 32..63)
    ir  = __builtin_amdgcn_wmma_f32_16x16x32_f16(false, aA1, false, load_fragA(wr, 32 + kb), (short)0, ir,  false, false);
    iz  = __builtin_amdgcn_wmma_f32_16x16x32_f16(false, aA1, false, load_fragA(wz, 32 + kb), (short)0, iz,  false, false);
    in_ = __builtin_amdgcn_wmma_f32_16x16x32_f16(false, aA1, false, load_fragA(wn, 32 + kb), (short)0, in_, false, false);
    hr  = __builtin_amdgcn_wmma_f32_16x16x32_f16(false, hA1, false, load_fragA(ur, 32 + kb), (short)0, hr,  false, false);
    hz  = __builtin_amdgcn_wmma_f32_16x16x32_f16(false, hA1, false, load_fragA(uz, 32 + kb), (short)0, hz,  false, false);
    hn  = __builtin_amdgcn_wmma_f32_16x16x32_f16(false, hA1, false, load_fragA(un, 32 + kb), (short)0, hn,  false, false);

    // fused gates; C/D layout: VGPR v holds row M = mrow0 + v, column N = m
    const float bir = b_ih[      j * 16 + m];
    const float biz = b_ih[ 64 + j * 16 + m];
    const float bin = b_ih[128 + j * 16 + m];
    const float bhr = b_hh[      j * 16 + m];
    const float bhz = b_hh[ 64 + j * 16 + m];
    const float bhn = b_hh[128 + j * 16 + m];
    #pragma unroll
    for (int v = 0; v < 8; ++v) {
      float r  = sigmoidf_(ir[v] + bir + hr[v] + bhr);
      float z  = sigmoidf_(iz[v] + biz + hz[v] + bhz);
      float nc = tanhf_(in_[v] + bin + r * (hn[v] + bhn));
      size_t idx = (size_t)(node0 + mrow0 + v) * H + j * 16 + m;
      float hold = hcur[idx];
      float hnew = (1.0f - z) * nc + z * hold;
      hnxt[idx] = hnew;
      h16[idx]  = (_Float16)hnew;  // tile-local rewrite; frags already consumed
    }
  }
}

// out = hT @ W_out + b_out   (OUT = 2)
__global__ __launch_bounds__(256) void head_kernel(const float* __restrict__ h,
                                                   const float* __restrict__ Wout,
                                                   const float* __restrict__ bout,
                                                   float* __restrict__ out, int N) {
  int i = blockIdx.x * 256 + threadIdx.x;
  if (i >= N) return;
  float a0 = bout[0], a1 = bout[1];
  const float* hr = h + (size_t)i * H;
  #pragma unroll
  for (int k = 0; k < H; ++k) {
    float hv = hr[k];
    a0 += hv * Wout[k * 2 + 0];
    a1 += hv * Wout[k * 2 + 1];
  }
  out[i * 2 + 0] = a0;
  out[i * 2 + 1] = a1;
}

// ---------------------------------------------------------------------------

static inline int cdiv(long long a, long long b) { return (int)((a + b - 1) / b); }

extern "C" void kernel_launch(void* const* d_in, const int* in_sizes, int n_in,
                              void* d_out, int out_size, void* d_ws, size_t ws_size,
                              hipStream_t stream) {
  const float* X    = (const float*)d_in[0];
  const int*   ei   = (const int*)  d_in[1];
  const float* Wg   = (const float*)d_in[2];
  const float* bg   = (const float*)d_in[3];
  const float* Wih  = (const float*)d_in[4];
  const float* Whh  = (const float*)d_in[5];
  const float* bih  = (const float*)d_in[6];
  const float* bhh  = (const float*)d_in[7];
  const float* Wout = (const float*)d_in[8];
  const float* bout = (const float*)d_in[9];

  const int N = in_sizes[0] / (T_STEPS * F_IN);
  const int E = in_sizes[1] / 2;
  const int* src = ei;
  const int* dst = ei + E;
  const size_t NH = (size_t)N * H;

  char* ws = (char*)d_ws;
  size_t off = 0;
  auto alloc = [&](size_t bytes) -> void* {
    void* p = ws + off;
    off = (off + bytes + 255) & ~(size_t)255;
    return p;
  };
  float*    deg   = (float*)   alloc(sizeof(float) * N);     // becomes dinv in place
  float*    wgt   = (float*)   alloc(sizeof(float) * E);
  float*    g     = (float*)   alloc(sizeof(float) * NH);
  float*    agg   = (float*)   alloc(sizeof(float) * NH);
  _Float16* a16   = (_Float16*)alloc(sizeof(_Float16) * NH);
  float*    hbufA = (float*)   alloc(sizeof(float) * NH);
  float*    hbufB = (float*)   alloc(sizeof(float) * NH);
  _Float16* h16   = (_Float16*)alloc(sizeof(_Float16) * NH);
  _Float16* Wih16 = (_Float16*)alloc(sizeof(_Float16) * 3 * H * H);
  _Float16* Whh16 = (_Float16*)alloc(sizeof(_Float16) * 3 * H * H);
  (void)ws_size;

  // ---- one-time graph preprocessing ----
  init_deg_kernel <<<cdiv(N, 256), 256, 0, stream>>>(deg, N);
  deg_accum_kernel<<<cdiv(E, 256), 256, 0, stream>>>(dst, deg, E);
  dinv_kernel     <<<cdiv(N, 256), 256, 0, stream>>>(deg, N);
  edge_w_kernel   <<<cdiv(E, 256), 256, 0, stream>>>(src, dst, deg, wgt, E);
  cvt_weights_kernel<<<cdiv(3 * H * H, 256), 256, 0, stream>>>(Wih, Whh, Wih16, Whh16, 3 * H * H);
  init_h_kernel   <<<cdiv((long long)NH, 256), 256, 0, stream>>>(hbufA, h16, (int)NH);

  float* hcur = hbufA;
  float* hnxt = hbufB;

  // ---- recurrent timesteps ----
  for (int t = 0; t < T_STEPS; ++t) {
    transform_kernel<<<cdiv(N, 4), 256, 0, stream>>>(X, Wg, deg, g, agg, t, N);
    scatter_kernel  <<<cdiv((long long)E * 16, 256), 256, 0, stream>>>(src, dst, wgt, g, agg, E);
    act_kernel      <<<cdiv((long long)NH, 256), 256, 0, stream>>>(agg, bg, a16, NH);
    gru_wmma_kernel <<<N / 16, 32, 0, stream>>>(a16, h16, hcur, hnxt, Wih16, Whh16, bih, bhh);
    float* tmp = hcur; hcur = hnxt; hnxt = tmp;
  }

  // ---- output head ----
  head_kernel<<<cdiv(N, 256), 256, 0, stream>>>(hcur, Wout, bout, (float*)d_out, N);
}
